// MultiHeadAttention_38182259262109
// MI455X (gfx1250) — matrix operand order repaired and checked by reading, not compiled, not run on hardware
//
#include <hip/hip_runtime.h>
#include <stdint.h>

typedef __attribute__((ext_vector_type(16))) __bf16 v16bf;
typedef __attribute__((ext_vector_type(8)))  float  v8f;

#define DEV __device__ __forceinline__

constexpr int Bb = 2, Tt = 2048, Ee = 1024, Hh = 16, Dd = 64;
constexpr int MT = Bb * Tt;               // 4096 rows of x
constexpr size_t HEAD_ELEMS = (size_t)Bb * Hh * Tt * Dd;  // 4 M bf16 elems = 8 MB

// ---------- bf16 packing: hardware v_cvt_pk_bf16_f32 ----------
DEV unsigned pack2(float a, float b) {
#if __has_builtin(__builtin_amdgcn_cvt_pk_bf16_f32)
    auto t = __builtin_amdgcn_cvt_pk_bf16_f32(a, b);
    unsigned u;
    __builtin_memcpy(&u, &t, 4);
    return u;
#else
    unsigned r;
    asm("v_cvt_pk_bf16_f32 %0, %1, %2" : "=v"(r) : "v"(a), "v"(b));
    return r;
#endif
}
DEV unsigned short f2bf(float f) {
    return (unsigned short)(pack2(f, f) & 0xffffu);
}

union AF { v16bf v; unsigned u[8]; };     // 16 bf16 = 8 dwords
union CF { v8f  v; float   f[8]; };

// A-frag (16x32 bf16) dword j -> K/2 offset within a 32-wide K tile:
//   lanes 0-15: dwords 0-7 hold K=0..15; lanes 16-31: K=16..31
DEV int a_khalf(int j, bool hi) { return j + (hi ? 8 : 0); }

// ---------------------------------------------------------------------------
// Kernel 1: QKV projection.  Y = x @ W^T, BK=64 (4 WMMAs per LDS stage).
// Q,K -> bf16 [B,H,T,D]; V -> bf16 [B,H,D,T] (transposed for the PV gemm).
// ---------------------------------------------------------------------------
__global__ __launch_bounds__(256)
void qkv_proj_kernel(const float* __restrict__ x,
                     const float* __restrict__ Wq, const float* __restrict__ Wk,
                     const float* __restrict__ Wv,
                     unsigned short* __restrict__ Q, unsigned short* __restrict__ K,
                     unsigned short* __restrict__ Vt)
{
    __shared__ unsigned short lA[64 * 64];   // A tile, row-major, stride 64
    __shared__ unsigned short lB[64 * 64];   // B tile, n-major: lB[n*64+k] = W[n0+n][k0+k]
    unsigned* lAu = (unsigned*)lA;
    unsigned* lBu = (unsigned*)lB;

    const int z    = blockIdx.z;
    const float* W = (z == 0) ? Wq : ((z == 1) ? Wk : Wv);
    const int row0 = blockIdx.x * 64;
    const int n0   = blockIdx.y * 64;
    const int tid  = threadIdx.x;
    const int lane = tid & 31;
    const int wv   = tid >> 5;          // 8 waves
    const int mstrip = wv >> 1;         // 0..3 : 16-row strip
    const int nh     = wv & 1;          // 0..1 : 32-col half
    const bool hi    = lane >= 16;
    const int  l15   = lane & 15;

    CF c0, c1;
#pragma unroll
    for (int i = 0; i < 8; i++) { c0.f[i] = 0.f; c1.f[i] = 0.f; }

    const int arow = tid >> 2;          // 0..63
    const int acg  = (tid & 3) * 16;    // 0,16,32,48

    for (int k0 = 0; k0 < Ee; k0 += 64) {
        // stage A (x) and B (W) as bf16: 16 floats each / thread
        const float4* xs = (const float4*)(x + (size_t)(row0 + arow) * Ee + k0 + acg);
        const float4* ws = (const float4*)(W + (size_t)(n0 + arow) * Ee + k0 + acg);
        if (k0 + 64 < Ee) {                      // prefetch next K tile
            __builtin_prefetch((const char*)(xs) + 256, 0, 3);
            __builtin_prefetch((const char*)(ws) + 256, 0, 3);
        }
#pragma unroll
        for (int h = 0; h < 2; h++) {
            float4 a0 = xs[2 * h], a1 = xs[2 * h + 1];
            uint4 pa;
            pa.x = pack2(a0.x, a0.y); pa.y = pack2(a0.z, a0.w);
            pa.z = pack2(a1.x, a1.y); pa.w = pack2(a1.z, a1.w);
            *(uint4*)&lAu[arow * 32 + (acg >> 1) + 4 * h] = pa;
            float4 b0 = ws[2 * h], b1 = ws[2 * h + 1];
            uint4 pb;
            pb.x = pack2(b0.x, b0.y); pb.y = pack2(b0.z, b0.w);
            pb.z = pack2(b1.x, b1.y); pb.w = pack2(b1.z, b1.w);
            *(uint4*)&lBu[arow * 32 + (acg >> 1) + 4 * h] = pb;
        }
        __syncthreads();

        AF a0, a1;
        const int mrow = mstrip * 16 + l15;
#pragma unroll
        for (int j = 0; j < 8; j++) {
            const int kh = a_khalf(j, hi);
            a0.u[j] = lAu[mrow * 32 + kh];
            a1.u[j] = lAu[mrow * 32 + 16 + kh];
        }
#pragma unroll
        for (int t = 0; t < 2; t++) {
            AF b0, b1;
            const int nn = nh * 32 + t * 16 + l15;
#pragma unroll
            for (int j = 0; j < 8; j++) {
                const int idx = j + (hi ? 8 : 0);
                b0.u[j] = lBu[nn * 32 + idx];
                b1.u[j] = lBu[nn * 32 + 16 + idx];
            }
            CF& c = t ? c1 : c0;
            c.v = __builtin_amdgcn_wmma_f32_16x16x32_bf16(false, a0.v, false, b0.v,
                                                          (short)0, c.v, false, false);
            c.v = __builtin_amdgcn_wmma_f32_16x16x32_bf16(false, a1.v, false, b1.v,
                                                          (short)0, c.v, false, false);
        }
        __syncthreads();
    }

    // epilogue: head-permuted bf16 stores
#pragma unroll
    for (int t = 0; t < 2; t++) {
        CF& c = t ? c1 : c0;
        const int coln = n0 + nh * 32 + t * 16 + l15;
        const int h = coln >> 6, d = coln & 63;
        if (z == 2) {
            // V transposed: this lane's 8 values are 8 consecutive trow slots
            const int grow0 = row0 + mstrip * 16 + (hi ? 8 : 0);
            const int bidx  = grow0 >> 11;          // / T
            const int trow0 = grow0 & (Tt - 1);     // multiple of 8 -> 16B aligned
            uint4 pv;
            pv.x = pack2(c.f[0], c.f[1]); pv.y = pack2(c.f[2], c.f[3]);
            pv.z = pack2(c.f[4], c.f[5]); pv.w = pack2(c.f[6], c.f[7]);
            *(uint4*)&Vt[((size_t)(bidx * Hh + h) * Dd + d) * Tt + trow0] = pv;
        } else {
            unsigned short* dst = (z == 0) ? Q : K;
#pragma unroll
            for (int r = 0; r < 8; r++) {
                const int grow = row0 + mstrip * 16 + r + (hi ? 8 : 0);
                const int bidx = grow >> 11;
                const int trow = grow & (Tt - 1);
                dst[((size_t)(bidx * Hh + h) * Tt + trow) * Dd + d] = f2bf(c.f[r]);
            }
        }
    }
}

// ---------------------------------------------------------------------------
// Kernel 2: causal flash attention, one workgroup = (b*H+h, 64 query rows),
// 4 waves x 16 query rows each, per-wave causal loop bound.
// ---------------------------------------------------------------------------
__global__ __launch_bounds__(128)
void attn_kernel(const unsigned short* __restrict__ Q,
                 const unsigned short* __restrict__ Kd,
                 const unsigned short* __restrict__ Vt,
                 unsigned short* __restrict__ Y)
{
    __shared__ unsigned lp[4][256];     // per-wave 16x32 bf16 P tile (16 dwords/row)

    const int tid  = threadIdx.x;
    const int lane = tid & 31;
    const int wv   = tid >> 5;                  // 0..3
    const bool hi  = lane >= 16;
    const int  l15 = lane & 15;
    const int bh   = blockIdx.y;                // b*H + h
    const int qb   = blockIdx.x * 64 + wv * 16; // first query row of this wave
    const int headT = bh * Tt;                  // row base into Q/K  [B,H,T,D]
    const int vBase = bh * Dd;                  // row base into Vt   [B,H,D,T]

    const unsigned* Qu = (const unsigned*)Q;    // 32 dwords per row
    const unsigned* Ku = (const unsigned*)Kd;
    const unsigned* Vu = (const unsigned*)Vt;   // 1024 dwords per row

    // Q A-frags (d = 0..31 and 32..63), kept in registers for the whole loop
    AF aq0, aq1;
    {
        const int qrow = headT + qb + l15;
#pragma unroll
        for (int j = 0; j < 8; j++) {
            const int kh = a_khalf(j, hi);
            aq0.u[j] = Qu[qrow * 32 + kh];
            aq1.u[j] = Qu[qrow * 32 + 16 + kh];
        }
    }

    CF o[4];
#pragma unroll
    for (int t = 0; t < 4; t++)
#pragma unroll
        for (int i = 0; i < 8; i++) o[t].f[i] = 0.f;

    float mrow[8], lrow[8];
#pragma unroll
    for (int r = 0; r < 8; r++) { mrow[r] = -1e30f; lrow[r] = 0.f; }

    const float scale = 0.125f;                 // 1/sqrt(64)
    const int jend = qb + 15;                   // causal bound for this wave

    for (int j0 = 0; j0 <= jend; j0 += 32) {
        // ---- S = Q K^T for 32 key columns (two 16-wide tiles) ----
        CF s[2];
#pragma unroll
        for (int jt = 0; jt < 2; jt++) {
            AF bk0, bk1;
            const int krow = headT + j0 + jt * 16 + l15;
#pragma unroll
            for (int i = 0; i < 8; i++) {
                const int idx = i + (hi ? 8 : 0);
                bk0.u[i] = Ku[krow * 32 + idx];
                bk1.u[i] = Ku[krow * 32 + 16 + idx];
            }
            v8f acc = {0.f, 0.f, 0.f, 0.f, 0.f, 0.f, 0.f, 0.f};
            acc = __builtin_amdgcn_wmma_f32_16x16x32_bf16(false, aq0.v, false, bk0.v,
                                                          (short)0, acc, false, false);
            acc = __builtin_amdgcn_wmma_f32_16x16x32_bf16(false, aq1.v, false, bk1.v,
                                                          (short)0, acc, false, false);
            s[jt].v = acc;
            const int jg = j0 + jt * 16 + l15;
#pragma unroll
            for (int r = 0; r < 8; r++) {
                const int qg = qb + r + (hi ? 8 : 0);
                const float val = s[jt].f[r] * scale;
                s[jt].f[r] = (jg > qg) ? -1e30f : val;   // v_cndmask, EXEC untouched
            }
        }

        // ---- online softmax: row lives across 16 lanes of one half ----
        float alpha[8];
#pragma unroll
        for (int r = 0; r < 8; r++) {
            float t = fmaxf(s[0].f[r], s[1].f[r]);
#pragma unroll
            for (int off = 1; off < 16; off <<= 1) t = fmaxf(t, __shfl_xor(t, off, 32));
            const float mn = fmaxf(mrow[r], t);
            alpha[r] = __expf(mrow[r] - mn);
            mrow[r]  = mn;
            const float p0 = __expf(s[0].f[r] - mn);
            const float p1 = __expf(s[1].f[r] - mn);
            s[0].f[r] = p0; s[1].f[r] = p1;
            float rs = p0 + p1;
#pragma unroll
            for (int off = 1; off < 16; off <<= 1) rs += __shfl_xor(rs, off, 32);
            lrow[r] = lrow[r] * alpha[r] + rs;
        }
#pragma unroll
        for (int t = 0; t < 4; t++)
#pragma unroll
            for (int r = 0; r < 8; r++) o[t].f[r] *= alpha[r];

        // ---- restage P (C-layout -> A-layout) through per-wave LDS ----
        unsigned short* lpw = (unsigned short*)lp[wv];   // 16 rows x 32 bf16
#pragma unroll
        for (int r = 0; r < 8; r++) {
            const int m = r + (hi ? 8 : 0);
            lpw[m * 32 + l15]      = f2bf(s[0].f[r]);
            lpw[m * 32 + 16 + l15] = f2bf(s[1].f[r]);
        }
        asm volatile("s_wait_dscnt 0" ::: "memory");     // wave-local DS flush

        AF pf;
#pragma unroll
        for (int j = 0; j < 8; j++) pf.u[j] = lp[wv][l15 * 16 + a_khalf(j, hi)];

        // ---- O += P V  (V transposed: B-frag contiguous per lane) ----
#pragma unroll
        for (int dt = 0; dt < 4; dt++) {
            AF bv;
            const int vrow = vBase + dt * 16 + l15;
#pragma unroll
            for (int i = 0; i < 8; i++)
                bv.u[i] = Vu[vrow * 1024 + (j0 >> 1) + i + (hi ? 8 : 0)];
            o[dt].v = __builtin_amdgcn_wmma_f32_16x16x32_bf16(false, pf.v, false, bv.v,
                                                              (short)0, o[dt].v, false, false);
        }
    }

    // ---- normalize and store concat-head bf16 row-major [B*T, E] ----
    const int b = bh >> 4, h = bh & 15;
#pragma unroll
    for (int r = 0; r < 8; r++) {
        const float inv = 1.f / lrow[r];
        const int trow = qb + r + (hi ? 8 : 0);
        const size_t yb = (size_t)(b * Tt + trow) * Ee + h * Dd;
#pragma unroll
        for (int dt = 0; dt < 4; dt++)
            Y[yb + dt * 16 + l15] = f2bf(o[dt].f[r] * inv);
    }
}

// ---------------------------------------------------------------------------
// Kernel 3: out = Y @ Wp^T + bp.  A tile (already bf16) staged with
// global_load_async_to_lds_b128 (ASYNCcnt path); BK=64.
// ---------------------------------------------------------------------------
__global__ __launch_bounds__(256)
void out_proj_kernel(const unsigned short* __restrict__ Yh,
                     const float* __restrict__ Wp, const float* __restrict__ bp,
                     float* __restrict__ out)
{
    __shared__ unsigned short lA[64 * 64];
    __shared__ unsigned short lB[64 * 64];
    unsigned* lAu = (unsigned*)lA;
    unsigned* lBu = (unsigned*)lB;

    const int row0 = blockIdx.x * 64;
    const int n0   = blockIdx.y * 64;
    const int tid  = threadIdx.x;
    const int lane = tid & 31;
    const int wv   = tid >> 5;
    const int mstrip = wv >> 1;
    const int nh     = wv & 1;
    const bool hi    = lane >= 16;
    const int  l15   = lane & 15;

    CF c0, c1;
#pragma unroll
    for (int i = 0; i < 8; i++) { c0.f[i] = 0.f; c1.f[i] = 0.f; }

    const int arow = tid >> 2;
    const int acg  = (tid & 3) * 16;    // bf16 elements: 16 elems = 32 bytes / thread

    // per-lane LDS byte offset of this thread's A-tile slot (low 32 bits of
    // the generic LDS address == LDS byte offset per the aperture mapping)
    const unsigned ldsA = (unsigned)(uintptr_t)&lA[arow * 64 + acg];

    for (int k0 = 0; k0 < Ee; k0 += 64) {
        // ---- async-stage A tile (raw bf16 copy, 32 B/thread) ----
        const unsigned short* gsrc = Yh + (size_t)(row0 + arow) * Ee + k0 + acg;
        asm volatile("global_load_async_to_lds_b128 %0, %1, off"
                     :: "v"(ldsA), "v"(gsrc) : "memory");
        asm volatile("global_load_async_to_lds_b128 %0, %1, off offset:16"
                     :: "v"(ldsA), "v"(gsrc) : "memory");

        // ---- stage B tile (Wp fp32 -> bf16) ----
        const float4* ws = (const float4*)(Wp + (size_t)(n0 + arow) * Ee + k0 + acg);
        if (k0 + 64 < Ee) __builtin_prefetch((const char*)(ws) + 256, 0, 3);
#pragma unroll
        for (int h = 0; h < 2; h++) {
            float4 b0 = ws[2 * h], b1 = ws[2 * h + 1];
            uint4 pb;
            pb.x = pack2(b0.x, b0.y); pb.y = pack2(b0.z, b0.w);
            pb.z = pack2(b1.x, b1.y); pb.w = pack2(b1.z, b1.w);
            *(uint4*)&lBu[arow * 32 + (acg >> 1) + 4 * h] = pb;
        }
        asm volatile("s_wait_asynccnt 0" ::: "memory");   // own async loads done
        __syncthreads();                                  // all waves' loads visible

        AF a0, a1;
        const int mrow = mstrip * 16 + l15;
#pragma unroll
        for (int j = 0; j < 8; j++) {
            const int kh = a_khalf(j, hi);
            a0.u[j] = lAu[mrow * 32 + kh];
            a1.u[j] = lAu[mrow * 32 + 16 + kh];
        }
#pragma unroll
        for (int t = 0; t < 2; t++) {
            AF b0, b1;
            const int nn = nh * 32 + t * 16 + l15;
#pragma unroll
            for (int j = 0; j < 8; j++) {
                const int idx = j + (hi ? 8 : 0);
                b0.u[j] = lBu[nn * 32 + idx];
                b1.u[j] = lBu[nn * 32 + 16 + idx];
            }
            CF& c = t ? c1 : c0;
            c.v = __builtin_amdgcn_wmma_f32_16x16x32_bf16(false, a0.v, false, b0.v,
                                                          (short)0, c.v, false, false);
            c.v = __builtin_amdgcn_wmma_f32_16x16x32_bf16(false, a1.v, false, b1.v,
                                                          (short)0, c.v, false, false);
        }
        __syncthreads();
    }

#pragma unroll
    for (int t = 0; t < 2; t++) {
        CF& c = t ? c1 : c0;
        const int coln = n0 + nh * 32 + t * 16 + l15;
        const float bias = bp[coln];
#pragma unroll
        for (int r = 0; r < 8; r++) {
            const int grow = row0 + mstrip * 16 + r + (hi ? 8 : 0);
            out[(size_t)grow * Ee + coln] = c.f[r] + bias;
        }
    }
}

// ---------------------------------------------------------------------------
extern "C" void kernel_launch(void* const* d_in, const int* in_sizes, int n_in,
                              void* d_out, int out_size, void* d_ws, size_t ws_size,
                              hipStream_t stream)
{
    const float* x  = (const float*)d_in[0];
    const float* Wq = (const float*)d_in[1];
    const float* Wk = (const float*)d_in[2];
    const float* Wv = (const float*)d_in[3];
    const float* Wp = (const float*)d_in[4];
    const float* bp = (const float*)d_in[5];
    float* out = (float*)d_out;

    unsigned short* Q  = (unsigned short*)d_ws;
    unsigned short* K  = Q  + HEAD_ELEMS;
    unsigned short* Vt = K  + HEAD_ELEMS;
    unsigned short* Y  = Vt + HEAD_ELEMS;      // 32 MB total scratch

    qkv_proj_kernel<<<dim3(MT / 64, Ee / 64, 3), 256, 0, stream>>>(x, Wq, Wk, Wv, Q, K, Vt);
    attn_kernel<<<dim3(Tt / 64, Bb * Hh), 128, 0, stream>>>(Q, K, Vt, Y);
    out_proj_kernel<<<dim3(MT / 64, Ee / 64), 256, 0, stream>>>(Y, Wp, bp, out);
}
